// GVAE_39058432590116
// MI455X (gfx1250) — matrix operand order repaired and checked by reading, not compiled, not run on hardware
//
#include <hip/hip_runtime.h>
#include <hip/hip_bf16.h>
#include <math.h>

// ---------------------------------------------------------------------------
// Graph-transformer decoder for MI455X (gfx1250, wave32, WMMA).
// ALL GEMMs run a branch-free bf16 WMMA kernel (float4 loads, f32 accum).
// Odd shapes (K=12 embed, O=14/12 heads) are zero-padded to WMMA-friendly
// sizes by tiny pad kernels instead of using guarded loads.
// ---------------------------------------------------------------------------

typedef __attribute__((ext_vector_type(16))) __bf16 v16bf;
typedef __attribute__((ext_vector_type(8)))  float  v8f;

#define N_PRIM 24
#define HID    256
#define HEADS  8
#define BATCH  128
#define NROWS  (BATCH * N_PRIM)           // 3072 node rows
#define EROWS  (BATCH * N_PRIM * N_PRIM)  // 73728 edge rows

__device__ __forceinline__ float leakyf(float x) { return x >= 0.f ? x : 0.1f * x; }

// ---- fragment loaders (branch-free, float4 -> bf16) -----------------------
template <int ACTIN>
__device__ __forceinline__ v16bf load_a_frag(const float* __restrict__ p)
{
    // elems 0..15 <- p[0..15] (contiguous K, matches B fragment)
    const float4* q0 = (const float4*)p;
    const float4* q1 = (const float4*)(p + 8);
    float4 a = q0[0], b = q0[1], c = q1[0], d = q1[1];
    if (ACTIN) {
        a.x = leakyf(a.x); a.y = leakyf(a.y); a.z = leakyf(a.z); a.w = leakyf(a.w);
        b.x = leakyf(b.x); b.y = leakyf(b.y); b.z = leakyf(b.z); b.w = leakyf(b.w);
        c.x = leakyf(c.x); c.y = leakyf(c.y); c.z = leakyf(c.z); c.w = leakyf(c.w);
        d.x = leakyf(d.x); d.y = leakyf(d.y); d.z = leakyf(d.z); d.w = leakyf(d.w);
    }
    v16bf r;
    r[0]  = (__bf16)a.x; r[1]  = (__bf16)a.y; r[2]  = (__bf16)a.z; r[3]  = (__bf16)a.w;
    r[4]  = (__bf16)b.x; r[5]  = (__bf16)b.y; r[6]  = (__bf16)b.z; r[7]  = (__bf16)b.w;
    r[8]  = (__bf16)c.x; r[9]  = (__bf16)c.y; r[10] = (__bf16)c.z; r[11] = (__bf16)c.w;
    r[12] = (__bf16)d.x; r[13] = (__bf16)d.y; r[14] = (__bf16)d.z; r[15] = (__bf16)d.w;
    return r;
}

__device__ __forceinline__ v16bf load_b_frag(const float* __restrict__ p)
{
    // 16 contiguous K values of one weight row (output column)
    const float4* q = (const float4*)p;
    float4 a = q[0], b = q[1], c = q[2], d = q[3];
    v16bf r;
    r[0]  = (__bf16)a.x; r[1]  = (__bf16)a.y; r[2]  = (__bf16)a.z; r[3]  = (__bf16)a.w;
    r[4]  = (__bf16)b.x; r[5]  = (__bf16)b.y; r[6]  = (__bf16)b.z; r[7]  = (__bf16)b.w;
    r[8]  = (__bf16)c.x; r[9]  = (__bf16)c.y; r[10] = (__bf16)c.z; r[11] = (__bf16)c.w;
    r[12] = (__bf16)d.x; r[13] = (__bf16)d.y; r[14] = (__bf16)d.z; r[15] = (__bf16)d.w;
    return r;
}

// ------------------- fast GEMM: out = act(A @ W^T + b) (+res) --------------
// Preconditions: M%16==0, K%32==0, O%64==0.  One wave -> 16x64 strip.
template <int ACTIN, int ACTOUT, int HASRES>
__global__ __launch_bounds__(256)
void k_gemm_fast(const float* __restrict__ A, const float* __restrict__ W,
                 const float* __restrict__ bias, const float* __restrict__ res,
                 float* __restrict__ out, int M, int K, int O)
{
    const int wid    = (blockIdx.x * blockDim.x + threadIdx.x) >> 5;
    const int lane   = threadIdx.x & 31;
    const int tilesO = O >> 6;
    if (wid >= (M >> 4) * tilesO) return;
    const int tm  = wid / tilesO;
    const int to  = wid % tilesO;
    const int lid = lane & 15;   // M row (A) / N column (B,C)
    const int kh  = lane >> 4;   // K-half selector

    const float* Arow = A + (size_t)(tm * 16 + lid) * K + kh * 16;
    const float* W0   = W + (size_t)(to * 64 +  0 + lid) * K + kh * 16;
    const float* W1   = W + (size_t)(to * 64 + 16 + lid) * K + kh * 16;
    const float* W2   = W + (size_t)(to * 64 + 32 + lid) * K + kh * 16;
    const float* W3   = W + (size_t)(to * 64 + 48 + lid) * K + kh * 16;

    const v8f zero = {0.f, 0.f, 0.f, 0.f, 0.f, 0.f, 0.f, 0.f};
    v8f acc0 = zero, acc1 = zero, acc2 = zero, acc3 = zero;

    for (int k0 = 0; k0 < K; k0 += 32) {
        v16bf af = load_a_frag<ACTIN>(Arow + k0);
        v16bf b0 = load_b_frag(W0 + k0);
        v16bf b1 = load_b_frag(W1 + k0);
        v16bf b2 = load_b_frag(W2 + k0);
        v16bf b3 = load_b_frag(W3 + k0);
        acc0 = __builtin_amdgcn_wmma_f32_16x16x32_bf16(false, af, false, b0,
                                                       (short)0, acc0, false, false);
        acc1 = __builtin_amdgcn_wmma_f32_16x16x32_bf16(false, af, false, b1,
                                                       (short)0, acc1, false, false);
        acc2 = __builtin_amdgcn_wmma_f32_16x16x32_bf16(false, af, false, b2,
                                                       (short)0, acc2, false, false);
        acc3 = __builtin_amdgcn_wmma_f32_16x16x32_bf16(false, af, false, b3,
                                                       (short)0, acc3, false, false);
    }

    // C/D layout: lane(l%16)=col, VGPR r -> row kh*8+r
    const int rowBase = tm * 16 + kh * 8;
    #pragma unroll
    for (int t = 0; t < 4; ++t) {
        const v8f acc = (t == 0) ? acc0 : (t == 1) ? acc1 : (t == 2) ? acc2 : acc3;
        const int col = to * 64 + t * 16 + lid;
        const float bv = bias[col];
        #pragma unroll
        for (int r = 0; r < 8; ++r) {
            size_t idx = (size_t)(rowBase + r) * O + col;
            float vL = acc[r] + bv;
            if (ACTOUT) vL = leakyf(vL);
            if (HASRES) vL += res[idx];
            out[idx] = vL;
        }
    }
}

// ------------------- narrow GEMM: O%16==0, one 16x16 tile per wave ---------
__global__ __launch_bounds__(256)
void k_gemm_fast16(const float* __restrict__ A, const float* __restrict__ W,
                   const float* __restrict__ bias, float* __restrict__ out,
                   int M, int K, int O)
{
    const int wid    = (blockIdx.x * blockDim.x + threadIdx.x) >> 5;
    const int lane   = threadIdx.x & 31;
    const int tilesO = O >> 4;
    if (wid >= (M >> 4) * tilesO) return;
    const int tm  = wid / tilesO;
    const int to  = wid % tilesO;
    const int lid = lane & 15;
    const int kh  = lane >> 4;

    const float* Arow = A + (size_t)(tm * 16 + lid) * K + kh * 16;
    const float* W0   = W + (size_t)(to * 16 + lid) * K + kh * 16;

    const v8f zero = {0.f, 0.f, 0.f, 0.f, 0.f, 0.f, 0.f, 0.f};
    v8f acc = zero;
    for (int k0 = 0; k0 < K; k0 += 32) {
        v16bf af = load_a_frag<0>(Arow + k0);
        v16bf b0 = load_b_frag(W0 + k0);
        acc = __builtin_amdgcn_wmma_f32_16x16x32_bf16(false, af, false, b0,
                                                      (short)0, acc, false, false);
    }
    const int rowBase = tm * 16 + kh * 8;
    const int col = to * 16 + lid;
    const float bv = bias[col];
    #pragma unroll
    for (int r = 0; r < 8; ++r)
        out[(size_t)(rowBase + r) * O + col] = acc[r] + bv;
}

// ------------------- zero-padding repack: dst[Rd,Cd] <- src[Rs,Cs] ---------
__global__ void k_pad2d(const float* __restrict__ src, float* __restrict__ dst,
                        int Rs, int Cs, int Rd, int Cd)
{
    int tid = blockIdx.x * blockDim.x + threadIdx.x;
    if (tid >= Rd * Cd) return;
    int r = tid / Cd, c = tid - r * Cd;
    dst[tid] = (r < Rs && c < Cs) ? src[(size_t)r * Cs + c] : 0.f;
}

// ------------------- LayerNorm over 256, one wave per row ------------------
__global__ __launch_bounds__(256)
void k_layernorm(const float* __restrict__ in, const float* __restrict__ g,
                 const float* __restrict__ b, float* __restrict__ out,
                 int rows, int act)
{
    int wid  = (blockIdx.x * blockDim.x + threadIdx.x) >> 5;
    int lane = threadIdx.x & 31;
    if (wid >= rows) return;
    const float4* x4 = (const float4*)(in + (size_t)wid * HID + lane * 8);
    float4 u = x4[0], w = x4[1];
    float v[8] = {u.x, u.y, u.z, u.w, w.x, w.y, w.z, w.w};
    float s = 0.f;
    #pragma unroll
    for (int e = 0; e < 8; ++e) s += v[e];
    #pragma unroll
    for (int off = 16; off >= 1; off >>= 1) s += __shfl_xor(s, off, 32);
    float m = s * (1.f / HID);
    float q = 0.f;
    #pragma unroll
    for (int e = 0; e < 8; ++e) { float d = v[e] - m; q += d * d; }
    #pragma unroll
    for (int off = 16; off >= 1; off >>= 1) q += __shfl_xor(q, off, 32);
    float rstd = rsqrtf(q * (1.f / HID) + 1e-5f);
    float* o = out + (size_t)wid * HID + lane * 8;
    #pragma unroll
    for (int e = 0; e < 8; ++e) {
        int c   = lane * 8 + e;
        float y = (v[e] - m) * rstd * g[c] + b[c];
        if (act) y = leakyf(y);
        o[e] = y;
    }
}

// ------- attention edge update: ne = att*em + att + ea (ea pre-stored in ne)
//         also emits per-head row sums (pre-softmax scores). 1 wave / (b,i,j).
__global__ __launch_bounds__(256)
void k_attn_edges(const float* __restrict__ q, const float* __restrict__ kk,
                  const float* __restrict__ em, float* __restrict__ ne,
                  float* __restrict__ sraw)
{
    int wid  = (blockIdx.x * blockDim.x + threadIdx.x) >> 5;
    int lane = threadIdx.x & 31;
    if (wid >= EROWS) return;
    int j  = wid % N_PRIM;
    int bi = wid / N_PRIM;       // b*24 + i
    int b  = bi / N_PRIM;
    const float* qr = q  + (size_t)bi * HID + lane * 8;
    const float* kr = kk + (size_t)(b * N_PRIM + j) * HID + lane * 8;
    const float* er = em + (size_t)wid * HID + lane * 8;
    float*       nr = ne + (size_t)wid * HID + lane * 8;
    float s = 0.f;
    #pragma unroll
    for (int e = 0; e < 8; ++e) {
        float att = qr[e] * kr[e] * 0.0625f;   // 1/sqrt(HID)
        float nv  = att * er[e] + att + nr[e]; // nr holds ea
        nr[e] = nv;
        s += nv;
    }
    // head h spans 32 contiguous d = 4 lanes; reduce within 4-lane groups
    s += __shfl_xor(s, 1, 32);
    s += __shfl_xor(s, 2, 32);
    if ((lane & 3) == 0) sraw[(size_t)wid * HEADS + (lane >> 2)] = s;
}

// ------- softmax over key axis j (24), in-place; one thread per (b,i,h) ----
__global__ void k_attn_softmax(float* __restrict__ sc)
{
    int tid = blockIdx.x * blockDim.x + threadIdx.x;
    if (tid >= BATCH * N_PRIM * HEADS) return;
    int h = tid & 7, r = tid >> 3;   // r = b*24 + i
    float* base = sc + (size_t)r * N_PRIM * HEADS + h;
    float m = -1e30f;
    #pragma unroll
    for (int j = 0; j < N_PRIM; ++j) m = fmaxf(m, base[j * HEADS]);
    float ex[N_PRIM]; float s = 0.f;
    #pragma unroll
    for (int j = 0; j < N_PRIM; ++j) { ex[j] = __expf(base[j * HEADS] - m); s += ex[j]; }
    float inv = 1.f / s;
    #pragma unroll
    for (int j = 0; j < N_PRIM; ++j) base[j * HEADS] = ex[j] * inv;
}

// ------- wv = einsum('bnmh,bmhd->bnhd'); one thread per output element -----
__global__ void k_attn_wv(const float* __restrict__ sc, const float* __restrict__ v,
                          float* __restrict__ wv)
{
    int tid = blockIdx.x * blockDim.x + threadIdx.x;
    if (tid >= NROWS * HID) return;
    int hd = tid & (HID - 1);
    int r  = tid >> 8;              // b*24 + i
    int h  = hd >> 5;
    int b  = r / N_PRIM;
    const float* sb = sc + (size_t)r * N_PRIM * HEADS + h;
    const float* vb = v  + (size_t)(b * N_PRIM) * HID + hd;
    float acc = 0.f;
    #pragma unroll
    for (int j = 0; j < N_PRIM; ++j) acc += sb[j * HEADS] * vb[j * HID];
    wv[tid] = acc;
}

// ------- output heads (inputs live in stride-16 padded buffers) ------------
__global__ void k_head_nodes(const float* __restrict__ in, float* __restrict__ out,
                             int rows)
{
    int r = blockIdx.x * blockDim.x + threadIdx.x;
    if (r >= rows) return;
    const float* x = in + (size_t)r * 16;   // padded stride
    float*       o = out + (size_t)r * 14;
    o[0] = 1.f / (1.f + __expf(-x[0]));
    float m = -1e30f;
    #pragma unroll
    for (int i = 1; i < 6; ++i) m = fmaxf(m, x[i]);
    float e[5]; float s = 0.f;
    #pragma unroll
    for (int i = 0; i < 5; ++i) { e[i] = __expf(x[1 + i] - m); s += e[i]; }
    float inv = 1.f / s;
    #pragma unroll
    for (int i = 0; i < 5; ++i) o[1 + i] = e[i] * inv;
    #pragma unroll
    for (int i = 6; i < 14; ++i) o[i] = x[i];
}

__global__ void k_head_edges(const float* __restrict__ in, float* __restrict__ out,
                             int rows)
{
    int r = blockIdx.x * blockDim.x + threadIdx.x;
    if (r >= rows) return;
    const float* x = in + (size_t)r * 16;   // padded stride
    float*       o = out + (size_t)r * 12;
    #pragma unroll
    for (int g = 0; g < 3; ++g) {
        float m = -1e30f;
        #pragma unroll
        for (int i = 0; i < 4; ++i) m = fmaxf(m, x[g * 4 + i]);
        float e[4]; float s = 0.f;
        #pragma unroll
        for (int i = 0; i < 4; ++i) { e[i] = __expf(x[g * 4 + i] - m); s += e[i]; }
        float inv = 1.f / s;
        #pragma unroll
        for (int i = 0; i < 4; ++i) o[g * 4 + i] = e[i] * inv;
    }
}

// ---------------------------------------------------------------------------
extern "C" void kernel_launch(void* const* d_in, const int* in_sizes, int n_in,
                              void* d_out, int out_size, void* d_ws, size_t ws_size,
                              hipStream_t stream)
{
    auto F = [&](int i) { return (const float*)d_in[i]; };

    // ---- workspace carve-up (f32) ----
    float* ws = (float*)d_ws;
    size_t off = 0;
    auto alloc = [&](size_t n) { float* r = ws + off; off += n; return r; };
    float* nodes = alloc((size_t)NROWS * HID);
    float* nln   = alloc((size_t)NROWS * HID);
    float* qb    = alloc((size_t)NROWS * HID);
    float* kb    = alloc((size_t)NROWS * HID);
    float* vb    = alloc((size_t)NROWS * HID);
    float* wvb   = alloc((size_t)NROWS * HID);
    float* edges = alloc((size_t)EROWS * HID);
    float* eln   = alloc((size_t)EROWS * HID);
    float* emb   = alloc((size_t)EROWS * HID);
    float* neb   = alloc((size_t)EROWS * HID);
    float* sc    = alloc((size_t)EROWS * HEADS);
    float* x1    = alloc((size_t)BATCH * 3072);
    float* x2    = alloc((size_t)BATCH * 3072);   // == nodes reshaped [3072,128]
    float* y1    = alloc((size_t)BATCH * 6912);
    float* e12   = alloc((size_t)BATCH * 6912);   // == edges reshaped [73728,12]
    float* e12p  = alloc((size_t)EROWS * 32);     // K-padded edge features
    float* wep   = alloc((size_t)HID * 32);       // edge_tf W0 padded [256,32]
    float* wn1p  = alloc((size_t)16 * HID);       // out_nodes W1 padded [16,256]
    float* bn1p  = alloc((size_t)16);
    float* we1p  = alloc((size_t)16 * HID);       // out_edges W1 padded [16,256]
    float* be1p  = alloc((size_t)16);
    float* n14p  = alloc((size_t)NROWS * 16);     // padded head outputs
    float* e12op = alloc((size_t)EROWS * 16);

    auto gemm = [&](const float* A, const float* W, const float* bias,
                    const float* res, float* out, int M, int K, int O,
                    int actOut, int actIn) {
        int tiles  = (M >> 4) * (O >> 6);
        int blocks = (tiles + 7) >> 3;   // 8 waves / block
        dim3 g(blocks), t(256);
        if (res) {
            if (actIn) {
                if (actOut) k_gemm_fast<1,1,1><<<g,t,0,stream>>>(A,W,bias,res,out,M,K,O);
                else        k_gemm_fast<1,0,1><<<g,t,0,stream>>>(A,W,bias,res,out,M,K,O);
            } else {
                if (actOut) k_gemm_fast<0,1,1><<<g,t,0,stream>>>(A,W,bias,res,out,M,K,O);
                else        k_gemm_fast<0,0,1><<<g,t,0,stream>>>(A,W,bias,res,out,M,K,O);
            }
        } else {
            if (actIn) {
                if (actOut) k_gemm_fast<1,1,0><<<g,t,0,stream>>>(A,W,bias,res,out,M,K,O);
                else        k_gemm_fast<1,0,0><<<g,t,0,stream>>>(A,W,bias,res,out,M,K,O);
            } else {
                if (actOut) k_gemm_fast<0,1,0><<<g,t,0,stream>>>(A,W,bias,res,out,M,K,O);
                else        k_gemm_fast<0,0,0><<<g,t,0,stream>>>(A,W,bias,res,out,M,K,O);
            }
        }
    };
    auto gemm16 = [&](const float* A, const float* W, const float* bias,
                      float* out, int M, int K, int O) {
        int tiles  = (M >> 4) * (O >> 4);
        int blocks = (tiles + 7) >> 3;
        k_gemm_fast16<<<dim3(blocks), dim3(256), 0, stream>>>(A, W, bias, out, M, K, O);
    };
    auto ln = [&](const float* in, const float* g, const float* b, float* out,
                  int rows, int act) {
        int blocks = (rows + 7) >> 3;
        k_layernorm<<<dim3(blocks), dim3(256), 0, stream>>>(in, g, b, out, rows, act);
    };
    auto pad = [&](const float* src, float* dst, int Rs, int Cs, int Rd, int Cd) {
        int n = Rd * Cd;
        k_pad2d<<<dim3((n + 255) / 256), dim3(256), 0, stream>>>(src, dst, Rs, Cs, Rd, Cd);
    };

    // ---- pad odd-shaped weights once (small) ----
    pad(F(13),  wep,  HID, 12,  HID, 32);   // edge_tf W0 [256,12] -> [256,32]
    pad(F(199), wn1p, 14,  HID, 16,  HID);  // out_nodes W1 [14,256] -> [16,256]
    pad(F(200), bn1p, 14,  1,   16,  1);
    pad(F(203), we1p, 12,  HID, 16,  HID);  // out_edges W1 [12,256] -> [16,256]
    pad(F(204), be1p, 12,  1,   16,  1);

    // ---- front-end: create_nodes / create_edges / embed to HID ----
    gemm(F(0), F(1),  F(2),  nullptr, x1,    BATCH, 512,  3072, 1, 0);
    gemm(x1,   F(3),  F(4),  nullptr, x2,    BATCH, 3072, 3072, 1, 0);
    gemm(x2,   F(9),  F(10), nullptr, nln,   NROWS, 128,  256,  1, 0);
    gemm(nln,  F(11), F(12), nullptr, nodes, NROWS, 256,  256,  0, 0);
    gemm(F(0), F(5),  F(6),  nullptr, y1,    BATCH, 512,  6912, 1, 0);
    gemm(y1,   F(7),  F(8),  nullptr, e12,   BATCH, 6912, 6912, 1, 0);
    pad(e12, e12p, EROWS, 12, EROWS, 32);   // zero-pad K: 12 -> 32
    gemm(e12p, wep,  F(14), nullptr, eln,   EROWS, 32,   256,  1, 0);
    gemm(eln,  F(15), F(16), nullptr, edges, EROWS, 256,  256,  0, 0);

    // ---- 6 transformer layers ----
    for (int l = 0; l < 6; ++l) {
        int p = 17 + 30 * l;
        ln(nodes, F(p + 0), F(p + 1), nln, NROWS, 0);
        ln(edges, F(p + 2), F(p + 3), eln, EROWS, 0);
        gemm(nln, F(p + 4),  F(p + 5),  nullptr, qb,  NROWS, 256, 256, 0, 0);
        gemm(nln, F(p + 6),  F(p + 7),  nullptr, kb,  NROWS, 256, 256, 0, 0);
        gemm(nln, F(p + 8),  F(p + 9),  nullptr, vb,  NROWS, 256, 256, 0, 0);
        gemm(eln, F(p + 10), F(p + 11), nullptr, emb, EROWS, 256, 256, 0, 0);
        gemm(eln, F(p + 12), F(p + 13), nullptr, neb, EROWS, 256, 256, 0, 0); // ea
        k_attn_edges<<<dim3(EROWS / 8), dim3(256), 0, stream>>>(qb, kb, emb, neb, sc);
        k_attn_softmax<<<dim3((BATCH * N_PRIM * HEADS + 255) / 256),
                         dim3(256), 0, stream>>>(sc);
        k_attn_wv<<<dim3((NROWS * HID + 255) / 256), dim3(256), 0, stream>>>(sc, vb, wvb);
        // an = n_out(wv) + nodes ; ae = e_out(leaky(ne)) + edges (fused residuals)
        gemm(wvb, F(p + 14), F(p + 15), nodes, nodes, NROWS, 256, 256, 0, 0);
        gemm(neb, F(p + 16), F(p + 17), edges, edges, EROWS, 256, 256, 0, 1);
        // node MLP: hn = leaky(LN(an)); nodes = mlp(hn) + an
        ln(nodes, F(p + 18), F(p + 19), nln, NROWS, 1);
        gemm(nln, F(p + 22), F(p + 23), nullptr, qb,    NROWS, 256, 256, 1, 0);
        gemm(qb,  F(p + 24), F(p + 25), nodes,   nodes, NROWS, 256, 256, 0, 0);
        // edge MLP: he = LN(ae); edges = mlp(he) + ae
        ln(edges, F(p + 20), F(p + 21), eln, EROWS, 0);
        gemm(eln, F(p + 26), F(p + 27), nullptr, emb,   EROWS, 256, 256, 1, 0);
        gemm(emb, F(p + 28), F(p + 29), edges,   edges, EROWS, 256, 256, 0, 0);
    }

    // ---- output heads (second linears via padded O=16 WMMA) ----
    gemm(nodes, F(197), F(198), nullptr, nln, NROWS, 256, 256, 1, 0);
    gemm16(nln, wn1p, bn1p, n14p, NROWS, 256, 16);
    gemm(edges, F(201), F(202), nullptr, eln, EROWS, 256, 256, 1, 0);
    gemm16(eln, we1p, be1p, e12op, EROWS, 256, 16);

    float* outf = (float*)d_out;
    k_head_nodes<<<dim3((NROWS + 255) / 256), dim3(256), 0, stream>>>(n14p, outf, NROWS);
    k_head_edges<<<dim3((EROWS + 255) / 256), dim3(256), 0, stream>>>(
        e12op, outf + (size_t)NROWS * 14, EROWS);
}
